// SelfAttentionBlock_45243185496228
// MI455X (gfx1250) — matrix operand order repaired and checked by reading, not compiled, not run on hardware
//
#include <hip/hip_runtime.h>
#include <hip/hip_bf16.h>
#include <stdint.h>

// Problem constants (fixed by the reference setup_inputs()).
#define B_   8
#define C_   512
#define N_   4096     // H*W = 64*64
#define G_   32
#define CPG  16       // channels per group
#define NH   8
#define DH   64       // head dim
#define GN_EPS 1e-5f

// Workspace requirement: ~170 MB
//   coefA/coefB: 2 * 8*512*4            =  32 KB
//   wqkv bf16 :  1536*512*2             = 1.5 MB
//   wproj bf16:  512*512*2              = 0.5 MB
//   h_t bf16  :  8*4096*512*2           = 32 MB
//   qkv bf16  :  8*1536*4096*2          = 96 MB
//   ao_t bf16 :  8*4096*512*2           = 32 MB

typedef __attribute__((ext_vector_type(16))) __bf16 bf16x16;
typedef __attribute__((ext_vector_type(8)))  float  f32x8;

union Frag { bf16x16 v; uint32_t u[8]; };

__device__ __forceinline__ uint16_t f2bf(float x) {
  uint32_t u = __float_as_uint(x);
  uint32_t r = u + 0x7FFFu + ((u >> 16) & 1u);   // round-to-nearest-even
  return (uint16_t)(r >> 16);
}

// A-matrix 16x32 bf16 fragment from row-major [rows x ld] bf16 storage.
// Lane L holds row M=L%16; lanes<16: K=0..15, lanes>=16: K=16..31 (2 per VGPR).
__device__ __forceinline__ Frag load_a(const uint16_t* __restrict__ p, int ld,
                                       int row0, int k0, int lane) {
  Frag f;
  const uint16_t* rp = p + (size_t)(row0 + (lane & 15)) * ld + k0 + 16 * (lane >> 4);
  #pragma unroll
  for (int v = 0; v < 8; ++v) {
    int k = v << 1;
    f.u[v] = *(const uint32_t*)(rp + k);
  }
  return f;
}

// B-matrix 32x16 bf16 fragment; storage is row-major over N (p[n][k], stride ld).
// Lane L holds column N=L%16; lanes<16: K=0..15, lanes>=16: K=16..31 (2 per VGPR).
__device__ __forceinline__ Frag load_b(const uint16_t* __restrict__ p, int ld,
                                       int n0, int k0, int lane) {
  Frag f;
  const uint16_t* rp = p + (size_t)(n0 + (lane & 15)) * ld + k0 + 16 * (lane >> 4);
  #pragma unroll
  for (int v = 0; v < 8; ++v)
    f.u[v] = *(const uint32_t*)(rp + 2 * v);
  return f;
}

// B-matrix 32x16 fragment where storage is p[k][n] with row stride ks (e.g. V matrix).
__device__ __forceinline__ Frag load_b_strided(const uint16_t* __restrict__ p, size_t ks,
                                               int k0, int n0, int lane) {
  Frag f;
  const uint16_t* bp = p + (size_t)(k0 + 16 * (lane >> 4)) * ks + n0 + (lane & 15);
  #pragma unroll
  for (int v = 0; v < 8; ++v) {
    uint32_t lo = bp[(size_t)(2 * v)     * ks];
    uint32_t hi = bp[(size_t)(2 * v + 1) * ks];
    f.u[v] = lo | (hi << 16);
  }
  return f;
}

__device__ __forceinline__ f32x8 wmma_bf16(Frag a, Frag b, f32x8 c) {
  return __builtin_amdgcn_wmma_f32_16x16x32_bf16(false, a.v, false, b.v,
                                                 (short)0, c, false, false);
}

// ---------------- Kernel 1: GroupNorm statistics -> per-channel affine ----------------
__global__ __launch_bounds__(256) void gn_stats_kernel(
    const float* __restrict__ x, const float* __restrict__ scale,
    const float* __restrict__ bias, float* __restrict__ coefA,
    float* __restrict__ coefB) {
  int bg = blockIdx.x;               // b * 32 + g
  int b = bg >> 5, g = bg & 31;
  const float* p = x + ((size_t)b * C_ + g * CPG) * N_;   // 65536 contiguous floats
  const int M = CPG * N_;
  float s = 0.f, sq = 0.f;
  for (int i = threadIdx.x; i < M; i += 256) { float v = p[i]; s += v; sq += v * v; }
  __shared__ float ssum[256], ssq[256];
  ssum[threadIdx.x] = s; ssq[threadIdx.x] = sq;
  __syncthreads();
  for (int st = 128; st > 0; st >>= 1) {
    if ((int)threadIdx.x < st) {
      ssum[threadIdx.x] += ssum[threadIdx.x + st];
      ssq[threadIdx.x]  += ssq[threadIdx.x + st];
    }
    __syncthreads();
  }
  if (threadIdx.x < CPG) {
    float mean = ssum[0] * (1.0f / M);
    float var  = ssq[0] * (1.0f / M) - mean * mean;
    float inv  = rsqrtf(var + GN_EPS);
    int c = g * CPG + threadIdx.x;
    float a = inv * scale[c];
    coefA[b * C_ + c] = a;
    coefB[b * C_ + c] = bias[c] - mean * a;
  }
}

// ---------------- Kernel 2: weights fp32 -> bf16 ----------------
__global__ __launch_bounds__(256) void cvt_w_kernel(
    const float* __restrict__ wq, const float* __restrict__ wp,
    uint16_t* __restrict__ oq, uint16_t* __restrict__ op) {
  int i = blockIdx.x * 256 + threadIdx.x;
  if (i < 3 * C_ * C_) oq[i] = f2bf(wq[i]);
  if (i < C_ * C_)     op[i] = f2bf(wp[i]);
}

// ---------------- Kernel 3: apply GN, write transposed bf16 h_t[b,n,c] ----------------
__global__ __launch_bounds__(256) void gn_apply_kernel(
    const float* __restrict__ x, const float* __restrict__ coefA,
    const float* __restrict__ coefB, uint16_t* __restrict__ h_t) {
  size_t idx = (size_t)blockIdx.x * 256 + threadIdx.x;  // over B*C*N
  int n = (int)(idx & (N_ - 1));
  int c = (int)((idx >> 12) & (C_ - 1));
  int b = (int)(idx >> 21);
  float h = coefA[b * C_ + c] * x[idx] + coefB[b * C_ + c];
  h_t[((size_t)b * N_ + n) * C_ + c] = f2bf(h);
}

// ---------------- Kernel 4: QKV GEMM (1536 x 4096 per batch, K=512) ----------------
// 8 waves/block; each wave owns a 32(o) x 64(n) tile => 8 accums, 8 WMMA per K-step
// for 6 fragment loads (2 A + 4 B). Block tile: 128(o) x 128(n).
__global__ __launch_bounds__(256) void qkv_gemm_kernel(
    const uint16_t* __restrict__ wq, const uint16_t* __restrict__ h_t,
    const float* __restrict__ qkv_bias, uint16_t* __restrict__ qkv) {
  int blk = blockIdx.x;
  int b   = blk / (12 * 32);
  int rem = blk % (12 * 32);
  int ot = rem / 32, nt = rem % 32;
  int lane = threadIdx.x & 31, w = threadIdx.x >> 5;
  int o0 = ot * 128 + (w >> 1) * 32;
  int n0 = nt * 128 + (w & 1) * 64;
  const uint16_t* hb = h_t + (size_t)b * N_ * C_;
  f32x8 zero = {0.f, 0.f, 0.f, 0.f, 0.f, 0.f, 0.f, 0.f};
  f32x8 acc[2][4] = {{zero, zero, zero, zero}, {zero, zero, zero, zero}};
  for (int kb = 0; kb < C_; kb += 32) {
    Frag a0 = load_a(wq, C_, o0,      kb, lane);
    Frag a1 = load_a(wq, C_, o0 + 16, kb, lane);
    Frag b0 = load_b(hb, C_, n0,      kb, lane);
    Frag b1 = load_b(hb, C_, n0 + 16, kb, lane);
    Frag b2 = load_b(hb, C_, n0 + 32, kb, lane);
    Frag b3 = load_b(hb, C_, n0 + 48, kb, lane);
    acc[0][0] = wmma_bf16(a0, b0, acc[0][0]);
    acc[0][1] = wmma_bf16(a0, b1, acc[0][1]);
    acc[0][2] = wmma_bf16(a0, b2, acc[0][2]);
    acc[0][3] = wmma_bf16(a0, b3, acc[0][3]);
    acc[1][0] = wmma_bf16(a1, b0, acc[1][0]);
    acc[1][1] = wmma_bf16(a1, b1, acc[1][1]);
    acc[1][2] = wmma_bf16(a1, b2, acc[1][2]);
    acc[1][3] = wmma_bf16(a1, b3, acc[1][3]);
  }
  int hi = lane >> 4, nn = lane & 15;
  #pragma unroll
  for (int i = 0; i < 2; ++i)
    #pragma unroll
    for (int j = 0; j < 4; ++j)
      #pragma unroll
      for (int r = 0; r < 8; ++r) {
        int o = o0 + 16 * i + r + 8 * hi;
        int n = n0 + 16 * j + nn;
        qkv[((size_t)b * 1536 + o) * N_ + n] = f2bf(acc[i][j][r] + qkv_bias[o]);
      }
}

// ---------------- Kernel 5: channel attention per (b, head) ----------------
__global__ __launch_bounds__(256) void attn_kernel(
    const uint16_t* __restrict__ qkv, uint16_t* __restrict__ ao_t) {
  int bh = blockIdx.x; int b = bh >> 3, h = bh & 7;
  int lane = threadIdx.x & 31, w = threadIdx.x >> 5;
  const uint16_t* qp = qkv + ((size_t)b * 1536 +          h * DH) * N_;
  const uint16_t* kp = qkv + ((size_t)b * 1536 + C_   +   h * DH) * N_;
  const uint16_t* vp = qkv + ((size_t)b * 1536 + 2*C_ +   h * DH) * N_;
  __shared__ __align__(16) float    sS[DH][DH];
  __shared__ __align__(16) uint16_t sP[DH][DH];

  // Phase 1: S = (Q K^T) * dh^-0.5 ; M=N=64, K=4096. wave w -> mt=w/2, nt in {2*(w&1), +1}
  int mt  = w >> 1;
  int nt0 = (w & 1) * 2;
  f32x8 zero = {0.f, 0.f, 0.f, 0.f, 0.f, 0.f, 0.f, 0.f};
  f32x8 a0 = zero, a1 = zero;
  for (int kb = 0; kb < N_; kb += 32) {
    Frag A  = load_a(qp, N_, mt * 16,        kb, lane);
    Frag B0 = load_b(kp, N_, nt0 * 16,       kb, lane);
    Frag B1 = load_b(kp, N_, (nt0 + 1) * 16, kb, lane);
    a0 = wmma_bf16(A, B0, a0);
    a1 = wmma_bf16(A, B1, a1);
  }
  int hi = lane >> 4, nn = lane & 15;
  #pragma unroll
  for (int r = 0; r < 8; ++r) {
    sS[mt * 16 + r + 8 * hi][nt0 * 16 + nn]       = a0[r] * 0.125f;
    sS[mt * 16 + r + 8 * hi][(nt0 + 1) * 16 + nn] = a1[r] * 0.125f;
  }
  __syncthreads();

  // Phase 2: row softmax (64 rows, fp32), result to LDS as bf16.
  if (threadIdx.x < DH) {
    int row = threadIdx.x;
    float m = -1e30f;
    for (int j = 0; j < DH; ++j) m = fmaxf(m, sS[row][j]);
    float s = 0.f;
    for (int j = 0; j < DH; ++j) s += __expf(sS[row][j] - m);
    float inv = 1.f / s;
    for (int j = 0; j < DH; ++j) sP[row][j] = f2bf(__expf(sS[row][j] - m) * inv);
  }
  __syncthreads();

  // Phase 3: O = P V ; M=64, K=64, N=4096. Preload P fragments (reused 32x per wave).
  Frag Af[4][2];
  #pragma unroll
  for (int m = 0; m < 4; ++m)
    #pragma unroll
    for (int ks = 0; ks < 2; ++ks)
      Af[m][ks] = load_a(&sP[0][0], DH, m * 16, ks * 32, lane);

  for (int nt = w; nt < 256; nt += 8) {
    f32x8 acc[4] = {zero, zero, zero, zero};
    #pragma unroll
    for (int ks = 0; ks < 2; ++ks) {
      Frag Bf = load_b_strided(vp, (size_t)N_, ks * 32, nt * 16, lane);
      #pragma unroll
      for (int m = 0; m < 4; ++m) acc[m] = wmma_bf16(Af[m][ks], Bf, acc[m]);
    }
    #pragma unroll
    for (int m = 0; m < 4; ++m)
      #pragma unroll
      for (int r = 0; r < 8; ++r) {
        int c = h * DH + m * 16 + r + 8 * hi;
        int n = nt * 16 + nn;
        ao_t[((size_t)b * N_ + n) * C_ + c] = f2bf(acc[m][r]);
      }
  }
}

// ---------------- Kernel 6: projection GEMM + bias + residual ----------------
// Same 32x64 wave tiling as the QKV GEMM; block tile 128(o) x 128(n).
__global__ __launch_bounds__(256) void proj_kernel(
    const uint16_t* __restrict__ wp, const uint16_t* __restrict__ ao_t,
    const float* __restrict__ proj_bias, const float* __restrict__ x,
    float* __restrict__ out) {
  int blk = blockIdx.x;
  int b   = blk / (4 * 32);
  int rem = blk % (4 * 32);
  int ot = rem / 32, nt = rem % 32;
  int lane = threadIdx.x & 31, w = threadIdx.x >> 5;
  int o0 = ot * 128 + (w >> 1) * 32;
  int n0 = nt * 128 + (w & 1) * 64;
  const uint16_t* ab = ao_t + (size_t)b * N_ * C_;
  f32x8 zero = {0.f, 0.f, 0.f, 0.f, 0.f, 0.f, 0.f, 0.f};
  f32x8 acc[2][4] = {{zero, zero, zero, zero}, {zero, zero, zero, zero}};
  for (int kb = 0; kb < C_; kb += 32) {
    Frag a0 = load_a(wp, C_, o0,      kb, lane);
    Frag a1 = load_a(wp, C_, o0 + 16, kb, lane);
    Frag b0 = load_b(ab, C_, n0,      kb, lane);
    Frag b1 = load_b(ab, C_, n0 + 16, kb, lane);
    Frag b2 = load_b(ab, C_, n0 + 32, kb, lane);
    Frag b3 = load_b(ab, C_, n0 + 48, kb, lane);
    acc[0][0] = wmma_bf16(a0, b0, acc[0][0]);
    acc[0][1] = wmma_bf16(a0, b1, acc[0][1]);
    acc[0][2] = wmma_bf16(a0, b2, acc[0][2]);
    acc[0][3] = wmma_bf16(a0, b3, acc[0][3]);
    acc[1][0] = wmma_bf16(a1, b0, acc[1][0]);
    acc[1][1] = wmma_bf16(a1, b1, acc[1][1]);
    acc[1][2] = wmma_bf16(a1, b2, acc[1][2]);
    acc[1][3] = wmma_bf16(a1, b3, acc[1][3]);
  }
  int hi = lane >> 4, nn = lane & 15;
  #pragma unroll
  for (int i = 0; i < 2; ++i)
    #pragma unroll
    for (int j = 0; j < 4; ++j)
      #pragma unroll
      for (int r = 0; r < 8; ++r) {
        int o = o0 + 16 * i + r + 8 * hi;
        int n = n0 + 16 * j + nn;
        size_t idx = ((size_t)b * C_ + o) * N_ + n;
        out[idx] = x[idx] + acc[i][j][r] + proj_bias[o];
      }
}

extern "C" void kernel_launch(void* const* d_in, const int* in_sizes, int n_in,
                              void* d_out, int out_size, void* d_ws, size_t ws_size,
                              hipStream_t stream) {
  (void)in_sizes; (void)n_in; (void)out_size; (void)ws_size;
  const float* x     = (const float*)d_in[0];
  const float* gsc   = (const float*)d_in[1];
  const float* gbi   = (const float*)d_in[2];
  const float* qkvw  = (const float*)d_in[3];
  const float* qkvb  = (const float*)d_in[4];
  const float* projw = (const float*)d_in[5];
  const float* projb = (const float*)d_in[6];
  float* out = (float*)d_out;

  char* ws = (char*)d_ws;
  size_t off = 0;
  float*    coefA = (float*)(ws + off);    off += (size_t)B_ * C_ * 4;
  float*    coefB = (float*)(ws + off);    off += (size_t)B_ * C_ * 4;
  uint16_t* wq    = (uint16_t*)(ws + off); off += (size_t)3 * C_ * C_ * 2;
  uint16_t* wp    = (uint16_t*)(ws + off); off += (size_t)C_ * C_ * 2;
  uint16_t* h_t   = (uint16_t*)(ws + off); off += (size_t)B_ * N_ * C_ * 2;
  uint16_t* qkv   = (uint16_t*)(ws + off); off += (size_t)B_ * 3 * C_ * N_ * 2;
  uint16_t* ao_t  = (uint16_t*)(ws + off); off += (size_t)B_ * N_ * C_ * 2;

  gn_stats_kernel<<<B_ * G_, 256, 0, stream>>>(x, gsc, gbi, coefA, coefB);
  cvt_w_kernel<<<(3 * C_ * C_ + 255) / 256, 256, 0, stream>>>(qkvw, projw, wq, wp);
  gn_apply_kernel<<<(B_ * C_ * N_) / 256, 256, 0, stream>>>(x, coefA, coefB, h_t);
  qkv_gemm_kernel<<<B_ * 12 * 32, 256, 0, stream>>>(wq, h_t, qkvb, qkv);
  attn_kernel<<<B_ * NH, 256, 0, stream>>>(qkv, ao_t);
  proj_kernel<<<B_ * 4 * 32, 256, 0, stream>>>(wp, ao_t, projb, x, out);
}
